// Mamba__9740985827496
// MI455X (gfx1250) — matrix operand order repaired and checked by reading, not compiled, not run on hardware
//
#include <hip/hip_runtime.h>
#include <hip/hip_bf16.h>
#include <math.h>

// Problem constants (match reference)
#define B_   4
#define T_   512
#define IN_  257
#define DM_  512
#define L_   6
#define DI_  1024
#define DS_  16
#define DC_  4
#define DR_  32
#define M_   (B_*T_)      // 2048 token rows
#define KPAD_IN 288       // 257 padded up to multiple of 32

typedef __attribute__((ext_vector_type(16))) __bf16 bf16x16;
typedef __attribute__((ext_vector_type(8)))  __bf16 bf16x8;
typedef __attribute__((ext_vector_type(8)))  float  f32x8;

__device__ __forceinline__ __bf16 f2bf(float f) {
  union { float f; unsigned u; } in; in.f = f;
  unsigned r = in.u + 0x7FFFu + ((in.u >> 16) & 1u);   // round-to-nearest-even
  unsigned short h = (unsigned short)(r >> 16);
  __bf16 o; __builtin_memcpy(&o, &h, 2); return o;
}

// ---------------- f32 -> bf16 convert with K padding ----------------
__global__ void k_cvt_pad(const float* __restrict__ in, __bf16* __restrict__ out,
                          int rows, int kin, int kp) {
  long e = (long)blockIdx.x * blockDim.x + threadIdx.x;
  long tot = (long)rows * kp;
  if (e >= tot) return;
  int r = (int)(e / kp);
  int k = (int)(e - (long)r * kp);
  out[e] = (k < kin) ? f2bf(in[(long)r * kin + k]) : f2bf(0.f);
}

// ---------------- bf16 WMMA GEMM: C(M,N) = A(M,Kp) * W(N,Kp)^T ----------------
// Register-blocked: each wave computes a 32x64 tile (2 M-subtiles x 4 N-subtiles);
// per K-step: 2 A fragments + 4 B fragments feed 8 WMMAs (A reused across N,
// B reused across M). Block = 4 waves stacked over M -> 128x64 per block.
// Fragment layouts per CDNA5 ISA 7.12.2 (wave32):
//   A 16x32 bf16: lane m=l&15, khalf=l>>4; frag[0..7]=A[m][k0+8*kh .. +7],
//                 frag[8..15]=A[m][k0+16+8*kh .. +7]    (two 16B loads)
//   B 32x16 bf16: lane n=l&15, khalf=l>>4; frag[0..7]=W[n][k0+8*kh .. +7],
//                 frag[8..15]=W[n][k0+16+8*kh .. +7]    (two 16B loads)
//   C 16x16 f32:  vgpr r -> row (lane<16 ? r : 8+r), col = lane&15
// epi: 0 = none, 1 = relu(x+bias), 2 = softplus(x+bias)
__global__ void k_gemm_bf16(const __bf16* __restrict__ A, const __bf16* __restrict__ W,
                            float* __restrict__ C, __bf16* __restrict__ Cb,
                            const float* __restrict__ bias,
                            int N, int Kp, int lda, int ldw, int ldc, int epi) {
  int lane  = threadIdx.x & 31;
  int wave  = threadIdx.x >> 5;
  int l15   = lane & 15;
  int khalf = lane >> 4;
  int mbase = blockIdx.y * 128 + wave * 32;
  int nbase = blockIdx.x * 64;

  const __bf16* arow0 = A + (size_t)(mbase + l15) * lda + 8 * khalf;
  const __bf16* arow1 = arow0 + (size_t)16 * lda;
  const __bf16* wrow[4];
#pragma unroll
  for (int j = 0; j < 4; j++) {
    int n = nbase + j * 16 + l15;
    if (n >= N) n = N - 1;                 // clamp reads; stores bounds-checked
    wrow[j] = W + (size_t)n * ldw + 8 * khalf;
  }

  f32x8 acc[2][4];
#pragma unroll
  for (int i = 0; i < 2; i++)
#pragma unroll
    for (int j = 0; j < 4; j++)
      acc[i][j] = (f32x8){0.f, 0.f, 0.f, 0.f, 0.f, 0.f, 0.f, 0.f};

  for (int k0 = 0; k0 < Kp; k0 += 32) {
    bf16x8 a0lo = *(const bf16x8*)(arow0 + k0);
    bf16x8 a0hi = *(const bf16x8*)(arow0 + k0 + 16);
    bf16x8 a1lo = *(const bf16x8*)(arow1 + k0);
    bf16x8 a1hi = *(const bf16x8*)(arow1 + k0 + 16);
    bf16x16 a0 = __builtin_shufflevector(a0lo, a0hi, 0,1,2,3,4,5,6,7,8,9,10,11,12,13,14,15);
    bf16x16 a1 = __builtin_shufflevector(a1lo, a1hi, 0,1,2,3,4,5,6,7,8,9,10,11,12,13,14,15);
    bf16x16 bfrag[4];
#pragma unroll
    for (int j = 0; j < 4; j++)
      bfrag[j] = __builtin_shufflevector(*(const bf16x8*)(wrow[j] + k0), *(const bf16x8*)(wrow[j] + k0 + 16), 0,1,2,3,4,5,6,7,8,9,10,11,12,13,14,15);
#pragma unroll
    for (int j = 0; j < 4; j++) {
      acc[0][j] = __builtin_amdgcn_wmma_f32_16x16x32_bf16(
                      false, a0, false, bfrag[j], (short)0, acc[0][j], false, false);
      acc[1][j] = __builtin_amdgcn_wmma_f32_16x16x32_bf16(
                      false, a1, false, bfrag[j], (short)0, acc[1][j], false, false);
    }
  }

#pragma unroll
  for (int j = 0; j < 4; j++) {
    int ncol = nbase + j * 16 + l15;
    if (ncol >= N) continue;               // per-lane store guard only
    float bv = bias ? bias[ncol] : 0.f;
#pragma unroll
    for (int i = 0; i < 2; i++) {
      int mb = mbase + i * 16 + ((lane < 16) ? 0 : 8);
#pragma unroll
      for (int r = 0; r < 8; r++) {
        float v = acc[i][j][r];
        if (epi == 1)      { v = fmaxf(v + bv, 0.f); }
        else if (epi == 2) { v += bv; v = (v > 20.f) ? v : log1pf(expf(v)); }
        size_t idx = (size_t)(mb + r) * ldc + ncol;
        C[idx] = v;
        if (Cb) Cb[idx] = f2bf(v);
      }
    }
  }
}

// ---------------- LayerNorm over DM=512, 128 threads x 4 elems ----------------
// flags: 1=add residual input, 2=write (in[+res]) back to res, 4=relu after affine
__global__ void k_ln(const float* __restrict__ in, float* __restrict__ res,
                     const float* __restrict__ w, const float* __restrict__ b,
                     float* __restrict__ of, __bf16* __restrict__ ob, int flags) {
  __shared__ float s1[128], s2[128];
  int row = blockIdx.x, tid = threadIdx.x;
  float v[4]; float s = 0.f, sq = 0.f;
#pragma unroll
  for (int j = 0; j < 4; j++) {
    int c = tid + j * 128;
    float x = in[(long)row * DM_ + c];
    if (flags & 1) x += res[(long)row * DM_ + c];
    v[j] = x; s += x; sq += x * x;
  }
  s1[tid] = s; s2[tid] = sq; __syncthreads();
  for (int o = 64; o > 0; o >>= 1) {
    if (tid < o) { s1[tid] += s1[tid + o]; s2[tid] += s2[tid + o]; }
    __syncthreads();
  }
  float mu  = s1[0] * (1.f / DM_);
  float var = s2[0] * (1.f / DM_) - mu * mu;
  float rs  = rsqrtf(var + 1e-5f);
#pragma unroll
  for (int j = 0; j < 4; j++) {
    int c = tid + j * 128;
    if (flags & 2) res[(long)row * DM_ + c] = v[j];
    float o = (v[j] - mu) * rs * w[c] + b[c];
    if (flags & 4) o = fmaxf(o, 0.f);
    if (of) of[(long)row * DM_ + c] = o;
    if (ob) ob[(long)row * DM_ + c] = f2bf(o);
  }
}

// ---------------- depthwise causal conv (DC=4) + bias + SiLU ----------------
__global__ void k_conv_silu(const float* __restrict__ xz, const float* __restrict__ cw,
                            const float* __restrict__ cb,
                            float* __restrict__ xcf, __bf16* __restrict__ xcb) {
  long e = (long)blockIdx.x * blockDim.x + threadIdx.x;
  if (e >= (long)M_ * DI_) return;
  int d = (int)(e % DI_);
  long bt = e / DI_;
  int t = (int)(bt % T_);
  float acc = cb[d];
#pragma unroll
  for (int j = 0; j < DC_; j++) {
    int tt = t - (DC_ - 1) + j;
    if (tt >= 0) acc += cw[d * DC_ + j] * xz[(bt + (tt - t)) * (2 * DI_) + d];
  }
  float s = acc / (1.f + expf(-acc));   // SiLU
  xcf[e] = s; xcb[e] = f2bf(s);
}

// ---------------- selective-scan: per (b,d) recurrence over T ----------------
// 16 blocks x 256 threads; Bm/Cm broadcast via LDS each timestep.
__global__ void k_scan(const float* __restrict__ dbl, const float* __restrict__ dt,
                       const float* __restrict__ xcf, const float* __restrict__ xz,
                       const float* __restrict__ alog, const float* __restrict__ Dp,
                       __bf16* __restrict__ ybf) {
  __shared__ float shB[DS_], shC[DS_];
  int b = blockIdx.x >> 2;
  int d = ((blockIdx.x & 3) << 8) + threadIdx.x;
  float a[DS_], st[DS_];
#pragma unroll
  for (int n = 0; n < DS_; n++) { a[n] = -expf(alog[(long)d * DS_ + n]); st[n] = 0.f; }
  float Dv = Dp[d];
  for (int t = 0; t < T_; t++) {
    long row = (long)b * T_ + t;
    if (threadIdx.x < 2 * DS_) {
      float v = dbl[row * 64 + DR_ + threadIdx.x];
      if (threadIdx.x < DS_) shB[threadIdx.x] = v; else shC[threadIdx.x - DS_] = v;
    }
    __syncthreads();
    float dtv = dt[row * DI_ + d];
    float xcv = xcf[row * DI_ + d];
    float zv  = xz[row * (2 * DI_) + DI_ + d];
    float y = 0.f;
#pragma unroll
    for (int n = 0; n < DS_; n++) {
      st[n] = expf(dtv * a[n]) * st[n] + dtv * shB[n] * xcv;
      y += st[n] * shC[n];
    }
    y += Dv * xcv;
    y *= zv / (1.f + expf(-zv));        // * silu(z)
    ybf[row * DI_ + d] = f2bf(y);
    __syncthreads();
  }
}

static inline size_t alignup(size_t x) { return (x + 255) & ~(size_t)255; }

extern "C" void kernel_launch(void* const* d_in, const int* in_sizes, int n_in,
                              void* d_out, int out_size, void* d_ws, size_t ws_size,
                              hipStream_t stream) {
  (void)in_sizes; (void)n_in; (void)out_size; (void)ws_size;
  const float* x         = (const float*)d_in[0];
  const float* w_in      = (const float*)d_in[1];
  const float* ln_in_w   = (const float*)d_in[2];
  const float* ln_in_b   = (const float*)d_in[3];
  const float* norm_w    = (const float*)d_in[4];
  const float* norm_b    = (const float*)d_in[5];
  const float* in_proj_w = (const float*)d_in[6];
  const float* conv_w    = (const float*)d_in[7];
  const float* conv_b    = (const float*)d_in[8];
  const float* x_proj_w  = (const float*)d_in[9];
  const float* dt_proj_w = (const float*)d_in[10];
  const float* dt_proj_b = (const float*)d_in[11];
  const float* A_log     = (const float*)d_in[12];
  const float* Dp        = (const float*)d_in[13];
  const float* out_proj_w= (const float*)d_in[14];
  const float* normf_w   = (const float*)d_in[15];
  const float* normf_b   = (const float*)d_in[16];
  const float* w_out     = (const float*)d_in[17];
  const float* b_out     = (const float*)d_in[18];

  char* p = (char*)d_ws;
  size_t off = 0;
  auto take = [&](size_t bytes) -> char* { char* r = p + off; off = alignup(off + bytes); return r; };

  __bf16* xpad = (__bf16*)take((size_t)M_ * KPAD_IN * 2);
  __bf16* winb = (__bf16*)take((size_t)DM_ * KPAD_IN * 2);
  __bf16* ipwb = (__bf16*)take((size_t)L_ * 2 * DI_ * DM_ * 2);
  __bf16* xpwb = (__bf16*)take((size_t)L_ * 64 * DI_ * 2);
  __bf16* dpwb = (__bf16*)take((size_t)L_ * DI_ * DR_ * 2);
  __bf16* opwb = (__bf16*)take((size_t)L_ * DM_ * DI_ * 2);
  __bf16* wob  = (__bf16*)take((size_t)IN_ * DM_ * 2);
  float*  h    = (float*) take((size_t)M_ * DM_ * 4);
  float*  res  = (float*) take((size_t)M_ * DM_ * 4);
  __bf16* hnb  = (__bf16*)take((size_t)M_ * DM_ * 2);
  float*  xz   = (float*) take((size_t)M_ * 2 * DI_ * 4);
  float*  xcf  = (float*) take((size_t)M_ * DI_ * 4);
  __bf16* xcb  = (__bf16*)take((size_t)M_ * DI_ * 2);
  float*  dbl  = (float*) take((size_t)M_ * 64 * 4);
  __bf16* dblb = (__bf16*)take((size_t)M_ * 64 * 2);
  float*  dtb  = (float*) take((size_t)M_ * DI_ * 4);
  __bf16* ybf  = (__bf16*)take((size_t)M_ * DI_ * 2);

  auto cvt = [&](const float* in, __bf16* out, int rows, int kin, int kp) {
    long tot = (long)rows * kp;
    k_cvt_pad<<<dim3((unsigned)((tot + 255) / 256)), dim3(256), 0, stream>>>(in, out, rows, kin, kp);
  };
  auto gemm = [&](const __bf16* A, const __bf16* W, float* C, __bf16* Cb, const float* bias,
                  int N, int Kp, int lda, int ldw, int ldc, int epi) {
    dim3 g((unsigned)((N + 63) / 64), M_ / 128);
    k_gemm_bf16<<<g, dim3(128), 0, stream>>>(A, W, C, Cb, bias, N, Kp, lda, ldw, ldc, epi);
  };

  // Weight / activation conversions (bf16 for WMMA)
  cvt(x,          xpad, M_,            IN_, KPAD_IN);
  cvt(w_in,       winb, DM_,           IN_, KPAD_IN);
  cvt(in_proj_w,  ipwb, L_ * 2 * DI_,  DM_, DM_);
  cvt(x_proj_w,   xpwb, L_ * 64,       DI_, DI_);
  cvt(dt_proj_w,  dpwb, L_ * DI_,      DR_, DR_);
  cvt(out_proj_w, opwb, L_ * DM_,      DI_, DI_);
  cvt(w_out,      wob,  IN_,           DM_, DM_);

  // h = relu(LN(x @ w_in^T))
  gemm(xpad, winb, h, nullptr, nullptr, DM_, KPAD_IN, KPAD_IN, KPAD_IN, DM_, 0);
  k_ln<<<M_, 128, 0, stream>>>(h, nullptr, ln_in_w, ln_in_b, h, nullptr, /*relu*/4);

  for (int i = 0; i < L_; i++) {
    int fl = (i == 0) ? 2 : (1 | 2);   // res = h (i==0) or h+res, write res; hn -> bf16
    k_ln<<<M_, 128, 0, stream>>>(h, res, norm_w + (size_t)i * DM_, norm_b + (size_t)i * DM_,
                                 nullptr, hnb, fl);
    // xz = hn @ in_proj^T  (2048 x 2048 x 512)
    gemm(hnb, ipwb + (size_t)i * 2 * DI_ * DM_, xz, nullptr, nullptr,
         2 * DI_, DM_, DM_, DM_, 2 * DI_, 0);
    // xc = silu(conv(xz[:, :DI]) + b)
    k_conv_silu<<<(M_ * DI_ + 255) / 256, 256, 0, stream>>>(
        xz, conv_w + (size_t)i * DI_ * DC_, conv_b + (size_t)i * DI_, xcf, xcb);
    // dbl = xc @ x_proj^T  (2048 x 64 x 1024)
    gemm(xcb, xpwb + (size_t)i * 64 * DI_, dbl, dblb, nullptr, 64, DI_, DI_, DI_, 64, 0);
    // dt = softplus(dbl[:, :32] @ dt_proj^T + b)  (2048 x 1024 x 32)
    gemm(dblb, dpwb + (size_t)i * DI_ * DR_, dtb, nullptr, dt_proj_b + (size_t)i * DI_,
         DI_, DR_, 64, DR_, DI_, 2);
    // selective scan + gates -> y (bf16)
    k_scan<<<16, 256, 0, stream>>>(dbl, dtb, xcf, xz,
                                   A_log + (size_t)i * DI_ * DS_, Dp + (size_t)i * DI_, ybf);
    // h = y @ out_proj^T  (2048 x 512 x 1024)
    gemm(ybf, opwb + (size_t)i * DM_ * DI_, h, nullptr, nullptr, DM_, DI_, DI_, DI_, DM_, 0);
  }

  // hf = LN(h + res); out = relu(hf @ w_out^T + b_out)
  k_ln<<<M_, 128, 0, stream>>>(h, res, normf_w, normf_b, nullptr, hnb, 1);
  gemm(hnb, wob, (float*)d_out, nullptr, b_out, IN_, DM_, DM_, DM_, IN_, 1);
}
